// Transformer_66340064854586
// MI455X (gfx1250) — matrix operand order repaired and checked by reading, not compiled, not run on hardware
//
#include <hip/hip_runtime.h>
#include <math.h>

// ---------------- problem constants (match reference) ----------------
#define BATCH  2
#define LSEQ   1024        // LS == LT
#define DMODEL 512
#define NHEAD  8
#define DFFN   2048
#define NLAYER 6
#define VTGT   32000
#define HDIM   64          // DMODEL / NHEAD
#define BL     (BATCH * LSEQ)   // 2048 rows of activations

typedef __attribute__((ext_vector_type(16))) __bf16          v16bf;
typedef __attribute__((ext_vector_type(16))) unsigned short  v16u;
typedef __attribute__((ext_vector_type(8)))  unsigned short  v8u;
typedef __attribute__((ext_vector_type(8)))  float           v8f;

// round-to-nearest-even fp32 -> bf16 bits
__device__ __forceinline__ unsigned short f2bf(float f) {
    union { float f; unsigned u; } v; v.f = f;
    unsigned u = v.u;
    unsigned r = u + 0x7FFFu + ((u >> 16) & 1u);
    return (unsigned short)(r >> 16);
}
__device__ __forceinline__ unsigned long long pack4(float4 f) {
    return (unsigned long long)f2bf(f.x)
         | ((unsigned long long)f2bf(f.y) << 16)
         | ((unsigned long long)f2bf(f.z) << 32)
         | ((unsigned long long)f2bf(f.w) << 48);
}

// ======================================================================
// Batched GEMM: C = alpha * A(MxK) x B(KxN) [+bias] [relu]
// fp32 in HBM -> bf16 tiles in LDS -> v_wmma_f32_16x16x32_bf16 (f32 acc).
// BT=1: B stored row-major NxK (used for Q.K^T).
// Batch offsets: z = blockIdx.z, ptr += (z/Hh)*s?b + (z%Hh)*s?h.
// Block: 256 thr = 8 wave32; block tile 128x64 (BK=32);
// wave tile 32x32 -> 2 A-frags x 2 B-frags -> 4 WMMA / K-step.
// LDS: A [m][k] (frag = two 16B runs), B transposed [n][k]
// (frag = one 32B run) so all fragment reads are ds_load_b128.
// ======================================================================
template<int BT>
__global__ __launch_bounds__(256)
void k_gemm(const float* __restrict__ A, const float* __restrict__ Bm,
            const float* __restrict__ bias, float* __restrict__ C,
            int M, int N, int K, int lda, int ldb, int ldc,
            int Hh, long long sAb, long long sAh,
            long long sBb, long long sBh,
            long long sCb, long long sCh,
            float alpha, int relu)
{
    __shared__ alignas(16) unsigned short sA [128 * 32];  // [m][k]
    __shared__ alignas(16) unsigned short sBt[ 64 * 32];  // [n][k] (transposed)

    int z  = blockIdx.z;
    int zb = z / Hh, zh = z % Hh;
    A  += (size_t)zb * sAb + (size_t)zh * sAh;
    Bm += (size_t)zb * sBb + (size_t)zh * sBh;
    C  += (size_t)zb * sCb + (size_t)zh * sCh;

    const int tid  = threadIdx.x;
    const int wave = tid >> 5, lane = tid & 31;
    const int wm   = wave & 3;          // 4 wave-rows of 32
    const int wn   = wave >> 2;         // 2 wave-cols of 32
    const int tileM = blockIdx.y * 128, tileN = blockIdx.x * 64;
    const int l  = lane & 15;           // M-row (A) / N-col (B,C) within tile
    const int hi = lane >> 4;           // lane half

    v8f acc[2][2] = {};

    for (int k0 = 0; k0 < K; k0 += 32) {
        if (k0 + 32 < K) {  // hint next tiles toward the caches
            __builtin_prefetch(&A[(size_t)tileM * lda + k0 + 32], 0, 1);
            if (BT) __builtin_prefetch(&Bm[(size_t)tileN * ldb + k0 + 32], 0, 1);
            else    __builtin_prefetch(&Bm[(size_t)(k0 + 32) * ldb + tileN], 0, 1);
        }
        // ---- stage A tile 128x32: 1024 float4 groups, 4 per thread ----
        #pragma unroll
        for (int i = 0; i < 4; ++i) {
            int g = tid + i * 256;
            int r = g >> 3, c4 = (g & 7) * 4;
            float4 f = *(const float4*)(&A[(size_t)(tileM + r) * lda + k0 + c4]);
            *(unsigned long long*)(&sA[r * 32 + c4]) = pack4(f);
        }
        // ---- stage B tile 32x64 into transposed [n][k]: 512 float4 groups ----
        #pragma unroll
        for (int i = 0; i < 2; ++i) {
            int g = tid + i * 256;
            if (BT) {   // B is NxK: rows are n, contiguous in k -> direct store
                int r = g >> 3, c4 = (g & 7) * 4;   // r = n, c4 = k
                float4 f = *(const float4*)(&Bm[(size_t)(tileN + r) * ldb + k0 + c4]);
                *(unsigned long long*)(&sBt[r * 32 + c4]) = pack4(f);
            } else {    // B is KxN: rows are k -> transpose on store
                int r = g >> 4, c4 = (g & 15) * 4;  // r = k, c4 = n
                float4 f = *(const float4*)(&Bm[(size_t)(k0 + r) * ldb + tileN + c4]);
                sBt[(c4 + 0) * 32 + r] = f2bf(f.x);
                sBt[(c4 + 1) * 32 + r] = f2bf(f.y);
                sBt[(c4 + 2) * 32 + r] = f2bf(f.z);
                sBt[(c4 + 3) * 32 + r] = f2bf(f.w);
            }
        }
        __syncthreads();

        // ---- build fragments (contiguous k = hi*16 + s in both A and B, wave32) ----
        // A 16x32: lane l holds row, halves = K runs [hi*16..+7] and [hi*16+8..+7]
        v16bf af[2];
        #pragma unroll
        for (int s = 0; s < 2; ++s) {
            int row = wm * 32 + s * 16 + l;
            v8u lo = *(const v8u*)(&sA[row * 32 + hi * 16]);
            v8u h8 = *(const v8u*)(&sA[row * 32 + hi * 16 + 8]);
            v16u u = __builtin_shufflevector(lo, h8, 0,1,2,3,4,5,6,7,
                                                     8,9,10,11,12,13,14,15);
            af[s] = __builtin_bit_cast(v16bf, u);
        }
        // B 32x16: lane l holds col n, halves = K run [hi*16 .. hi*16+15] (contig)
        v16bf bfr[2];
        #pragma unroll
        for (int t = 0; t < 2; ++t) {
            int col = wn * 32 + t * 16 + l;
            v16u u = *(const v16u*)(&sBt[col * 32 + hi * 16]);
            bfr[t] = __builtin_bit_cast(v16bf, u);
        }

        acc[0][0] = __builtin_amdgcn_wmma_f32_16x16x32_bf16(
                        false, af[0], false, bfr[0], (short)0, acc[0][0], false, false);
        acc[0][1] = __builtin_amdgcn_wmma_f32_16x16x32_bf16(
                        false, af[0], false, bfr[1], (short)0, acc[0][1], false, false);
        acc[1][0] = __builtin_amdgcn_wmma_f32_16x16x32_bf16(
                        false, af[1], false, bfr[0], (short)0, acc[1][0], false, false);
        acc[1][1] = __builtin_amdgcn_wmma_f32_16x16x32_bf16(
                        false, af[1], false, bfr[1], (short)0, acc[1][1], false, false);
        __syncthreads();
    }

    // ---- epilogue: C/D f32 layout — VGPR r: row = r + hi*8, col = lane&15 ----
    #pragma unroll
    for (int s = 0; s < 2; ++s)
    #pragma unroll
    for (int t = 0; t < 2; ++t)
    #pragma unroll
    for (int r = 0; r < 8; ++r) {
        int gm = tileM + wm * 32 + s * 16 + hi * 8 + r;
        int gn = tileN + wn * 32 + t * 16 + l;
        float v = acc[s][t][r] * alpha;
        if (bias) v += bias[gn];
        if (relu) v = fmaxf(v, 0.f);
        C[(size_t)gm * ldc + gn] = v;
    }
}

// ---------------- embedding + sinusoidal positional encoding ----------------
__global__ __launch_bounds__(256)
void k_embed(const float* __restrict__ emb, const int* __restrict__ tok,
             float* __restrict__ out, int L, int Dm, float scale)
{
    int row = blockIdx.x;            // b*L + l
    int pos = row % L;
    int t   = tok[row];
    const float* e = emb + (size_t)t * Dm;
    float* o = out + (size_t)row * Dm;
    const float nlog = -9.210340371976184f / (float)Dm;  // -ln(10000)/D
    for (int d = threadIdx.x; d < Dm; d += blockDim.x) {
        int i2 = d & ~1;
        float ang = (float)pos * __expf((float)i2 * nlog);
        float pe  = (d & 1) ? __cosf(ang) : __sinf(ang);
        o[d] = e[d] * scale + pe;
    }
}

// ---------------- LayerNorm (torch-style: ddof=1, eps on std) ----------------
__global__ __launch_bounds__(256)
void k_ln(const float* __restrict__ x, const float* __restrict__ a,
          const float* __restrict__ b, float* __restrict__ o, int Dm)
{
    __shared__ float rs[256], rq[256];
    int row = blockIdx.x;
    const float* p = x + (size_t)row * Dm;
    float s = 0.f, q = 0.f;
    for (int d = threadIdx.x; d < Dm; d += 256) { float v = p[d]; s += v; q += v * v; }
    rs[threadIdx.x] = s; rq[threadIdx.x] = q;
    __syncthreads();
    for (int st = 128; st > 0; st >>= 1) {
        if (threadIdx.x < st) { rs[threadIdx.x] += rs[threadIdx.x + st];
                                rq[threadIdx.x] += rq[threadIdx.x + st]; }
        __syncthreads();
    }
    float mean = rs[0] / (float)Dm;
    float var  = (rq[0] - mean * rs[0]) / (float)(Dm - 1);
    float istd = 1.0f / (sqrtf(fmaxf(var, 0.f)) + 1e-5f);
    float* oo = o + (size_t)row * Dm;
    for (int d = threadIdx.x; d < Dm; d += 256)
        oo[d] = a[d] * (p[d] - mean) * istd + b[d];
}

// ---------------- masked softmax over score rows ----------------
__global__ __launch_bounds__(256)
void k_softmax(float* __restrict__ S, int Lq, int Lk, int causal)
{
    __shared__ float red[256];
    size_t row = blockIdx.x;                 // over B*H*Lq
    float* p = S + row * (size_t)Lk;
    int q   = (int)(row % (size_t)Lq);
    int lim = causal ? (q + 1) : Lk;
    float m = -3.0e38f;
    for (int k = threadIdx.x; k < lim; k += 256) m = fmaxf(m, p[k]);
    red[threadIdx.x] = m; __syncthreads();
    for (int st = 128; st > 0; st >>= 1) {
        if (threadIdx.x < st) red[threadIdx.x] = fmaxf(red[threadIdx.x], red[threadIdx.x + st]);
        __syncthreads();
    }
    m = red[0];
    __syncthreads();
    float s = 0.f;
    for (int k = threadIdx.x; k < Lk; k += 256) {
        float e = (k < lim) ? __expf(p[k] - m) : 0.f;
        p[k] = e; s += e;
    }
    red[threadIdx.x] = s; __syncthreads();
    for (int st = 128; st > 0; st >>= 1) {
        if (threadIdx.x < st) red[threadIdx.x] += red[threadIdx.x + st];
        __syncthreads();
    }
    float inv = 1.0f / red[0];
    for (int k = threadIdx.x; k < Lk; k += 256) p[k] *= inv;
}

// ---------------- residual add ----------------
__global__ void k_add(float* __restrict__ x, const float* __restrict__ y, int n)
{
    int i = blockIdx.x * blockDim.x + threadIdx.x;
    if (i < n) x[i] += y[i];
}

// ---------------- in-place log-softmax over vocab ----------------
__global__ __launch_bounds__(256)
void k_logsoftmax(float* __restrict__ X, int Nv)
{
    __shared__ float red[256];
    float* p = X + (size_t)blockIdx.x * Nv;
    float m = -3.0e38f;
    for (int i = threadIdx.x; i < Nv; i += 256) m = fmaxf(m, p[i]);
    red[threadIdx.x] = m; __syncthreads();
    for (int st = 128; st > 0; st >>= 1) {
        if (threadIdx.x < st) red[threadIdx.x] = fmaxf(red[threadIdx.x], red[threadIdx.x + st]);
        __syncthreads();
    }
    m = red[0];
    __syncthreads();
    float s = 0.f;
    for (int i = threadIdx.x; i < Nv; i += 256) s += __expf(p[i] - m);
    red[threadIdx.x] = s; __syncthreads();
    for (int st = 128; st > 0; st >>= 1) {
        if (threadIdx.x < st) red[threadIdx.x] += red[threadIdx.x + st];
        __syncthreads();
    }
    float lse = m + logf(red[0]);
    for (int i = threadIdx.x; i < Nv; i += 256) p[i] -= lse;
}

// ======================================================================
// host-side helpers
// ======================================================================
static inline void gemm(hipStream_t st, const float* A, const float* Bm,
                        const float* bias, float* C,
                        int M, int N, int K, int lda, int ldb, int ldc,
                        int batch, int Hh,
                        long long sAb, long long sAh,
                        long long sBb, long long sBh,
                        long long sCb, long long sCh,
                        float alpha, int relu, int bt)
{
    dim3 g(N / 64, M / 128, batch), b(256);
    if (bt) k_gemm<1><<<g, b, 0, st>>>(A, Bm, bias, C, M, N, K, lda, ldb, ldc,
                                       Hh, sAb, sAh, sBb, sBh, sCb, sCh, alpha, relu);
    else    k_gemm<0><<<g, b, 0, st>>>(A, Bm, bias, C, M, N, K, lda, ldb, ldc,
                                       Hh, sAb, sAh, sBb, sBh, sCb, sCh, alpha, relu);
}

static inline void resadd(hipStream_t st, float* x, const float* y) {
    k_add<<<(BL * DMODEL) / 256, 256, 0, st>>>(x, y, BL * DMODEL);
}

// full multi-head attention block: T = MHA(Hq as queries, Hkv as keys/values)
static inline void mha(hipStream_t st, const float* Hq, const float* Hkv,
                       const float* wq, const float* wk, const float* wv, const float* wo,
                       float* Qb, float* Kb, float* Vb, float* Sc, float* AO, float* T,
                       int causal)
{
    const long long sBD  = (long long)LSEQ * DMODEL;       // per-batch activation stride
    const long long sQK  = (long long)LSEQ * LSEQ;         // per-head score stride
    // projections (M=BL, N=K=D)
    gemm(st, Hq,  wq, nullptr, Qb, BL, DMODEL, DMODEL, DMODEL, DMODEL, DMODEL,
         1, 1, 0, 0, 0, 0, 0, 0, 1.f, 0, 0);
    gemm(st, Hkv, wk, nullptr, Kb, BL, DMODEL, DMODEL, DMODEL, DMODEL, DMODEL,
         1, 1, 0, 0, 0, 0, 0, 0, 1.f, 0, 0);
    gemm(st, Hkv, wv, nullptr, Vb, BL, DMODEL, DMODEL, DMODEL, DMODEL, DMODEL,
         1, 1, 0, 0, 0, 0, 0, 0, 1.f, 0, 0);
    // scores = (1/sqrt(HD)) * Q . K^T   batched over (b,h)
    gemm(st, Qb, Kb, nullptr, Sc, LSEQ, LSEQ, HDIM, DMODEL, DMODEL, LSEQ,
         BATCH * NHEAD, NHEAD,
         sBD, HDIM, sBD, HDIM, (long long)NHEAD * sQK, sQK,
         0.125f, 0, 1);
    k_softmax<<<BATCH * NHEAD * LSEQ, 256, 0, st>>>(Sc, LSEQ, LSEQ, causal);
    // out = attn . V   batched over (b,h); writes heads back into [BL, D]
    gemm(st, Sc, Vb, nullptr, AO, LSEQ, HDIM, LSEQ, LSEQ, DMODEL, DMODEL,
         BATCH * NHEAD, NHEAD,
         (long long)NHEAD * sQK, sQK, sBD, HDIM, sBD, HDIM,
         1.f, 0, 0);
    // output projection
    gemm(st, AO, wo, nullptr, T, BL, DMODEL, DMODEL, DMODEL, DMODEL, DMODEL,
         1, 1, 0, 0, 0, 0, 0, 0, 1.f, 0, 0);
}

extern "C" void kernel_launch(void* const* d_in, const int* in_sizes, int n_in,
                              void* d_out, int out_size, void* d_ws, size_t ws_size,
                              hipStream_t stream)
{
    (void)in_sizes; (void)n_in; (void)out_size; (void)ws_size;
    const float* src_emb  = (const float*)d_in[0];
    const float* tgt_emb  = (const float*)d_in[1];
    const float* enc_wq   = (const float*)d_in[2];
    const float* enc_wk   = (const float*)d_in[3];
    const float* enc_wv   = (const float*)d_in[4];
    const float* enc_wo   = (const float*)d_in[5];
    const float* enc_w1   = (const float*)d_in[6];
    const float* enc_b1   = (const float*)d_in[7];
    const float* enc_w2   = (const float*)d_in[8];
    const float* enc_b2   = (const float*)d_in[9];
    const float* enc_ln_a = (const float*)d_in[10];
    const float* enc_ln_b = (const float*)d_in[11];
    const float* enc_fa   = (const float*)d_in[12];
    const float* enc_fb   = (const float*)d_in[13];
    const float* dec_sq   = (const float*)d_in[14];
    const float* dec_sk   = (const float*)d_in[15];
    const float* dec_sv   = (const float*)d_in[16];
    const float* dec_so   = (const float*)d_in[17];
    const float* dec_cq   = (const float*)d_in[18];
    const float* dec_ck   = (const float*)d_in[19];
    const float* dec_cv   = (const float*)d_in[20];
    const float* dec_co   = (const float*)d_in[21];
    const float* dec_w1   = (const float*)d_in[22];
    const float* dec_b1   = (const float*)d_in[23];
    const float* dec_w2   = (const float*)d_in[24];
    const float* dec_b2   = (const float*)d_in[25];
    const float* dec_ln_a = (const float*)d_in[26];
    const float* dec_ln_b = (const float*)d_in[27];
    const float* dec_fa   = (const float*)d_in[28];
    const float* dec_fb   = (const float*)d_in[29];
    const float* proj_w   = (const float*)d_in[30];
    const float* proj_b   = (const float*)d_in[31];
    const int*   src_tok  = (const int*)d_in[32];
    const int*   tgt_tok  = (const int*)d_in[33];

    // ----- workspace layout (~122 MB fp32) -----
    float* ws  = (float*)d_ws;
    const size_t ACT = (size_t)BL * DMODEL;          // 1,048,576 floats
    float* X   = ws; ws += ACT;                      // encoder stream
    float* Y   = ws; ws += ACT;                      // decoder stream
    float* Hb  = ws; ws += ACT;                      // LN output
    float* Qb  = ws; ws += ACT;
    float* Kb  = ws; ws += ACT;
    float* Vb  = ws; ws += ACT;
    float* AO  = ws; ws += ACT;
    float* T   = ws; ws += ACT;
    float* ENC = ws; ws += ACT;                      // encoder output
    float* FF  = ws; ws += (size_t)BL * DFFN;        // 4,194,304 floats
    float* Sc  = ws; ws += (size_t)BATCH * NHEAD * LSEQ * LSEQ;  // 16,777,216

    const float scale = 22.627416997969522f;         // sqrt(512)
    const size_t DD = (size_t)DMODEL * DMODEL;
    const size_t DF = (size_t)DMODEL * DFFN;

    // ===================== encoder =====================
    k_embed<<<BL, 256, 0, stream>>>(src_emb, src_tok, X, LSEQ, DMODEL, scale);
    for (int i = 0; i < NLAYER; ++i) {
        const float* wq = enc_wq + i * DD;
        const float* wk = enc_wk + i * DD;
        const float* wv = enc_wv + i * DD;
        const float* wo = enc_wo + i * DD;
        const float* a0 = enc_ln_a + ((size_t)i * 2 + 0) * DMODEL;
        const float* b0 = enc_ln_b + ((size_t)i * 2 + 0) * DMODEL;
        const float* a1 = enc_ln_a + ((size_t)i * 2 + 1) * DMODEL;
        const float* b1l= enc_ln_b + ((size_t)i * 2 + 1) * DMODEL;

        k_ln<<<BL, 256, 0, stream>>>(X, a0, b0, Hb, DMODEL);
        mha(stream, Hb, Hb, wq, wk, wv, wo, Qb, Kb, Vb, Sc, AO, T, /*causal=*/0);
        resadd(stream, X, T);

        k_ln<<<BL, 256, 0, stream>>>(X, a1, b1l, Hb, DMODEL);
        gemm(stream, Hb, enc_w1 + i * DF, enc_b1 + (size_t)i * DFFN, FF,
             BL, DFFN, DMODEL, DMODEL, DFFN, DFFN,
             1, 1, 0, 0, 0, 0, 0, 0, 1.f, /*relu=*/1, 0);
        gemm(stream, FF, enc_w2 + i * DF, enc_b2 + (size_t)i * DMODEL, T,
             BL, DMODEL, DFFN, DFFN, DMODEL, DMODEL,
             1, 1, 0, 0, 0, 0, 0, 0, 1.f, 0, 0);
        resadd(stream, X, T);
    }
    k_ln<<<BL, 256, 0, stream>>>(X, enc_fa, enc_fb, ENC, DMODEL);

    // ===================== decoder =====================
    k_embed<<<BL, 256, 0, stream>>>(tgt_emb, tgt_tok, Y, LSEQ, DMODEL, scale);
    for (int i = 0; i < NLAYER; ++i) {
        const float* a0 = dec_ln_a + ((size_t)i * 3 + 0) * DMODEL;
        const float* b0 = dec_ln_b + ((size_t)i * 3 + 0) * DMODEL;
        const float* a1 = dec_ln_a + ((size_t)i * 3 + 1) * DMODEL;
        const float* b1l= dec_ln_b + ((size_t)i * 3 + 1) * DMODEL;
        const float* a2 = dec_ln_a + ((size_t)i * 3 + 2) * DMODEL;
        const float* b2l= dec_ln_b + ((size_t)i * 3 + 2) * DMODEL;

        // causal self-attention
        k_ln<<<BL, 256, 0, stream>>>(Y, a0, b0, Hb, DMODEL);
        mha(stream, Hb, Hb, dec_sq + i * DD, dec_sk + i * DD, dec_sv + i * DD,
            dec_so + i * DD, Qb, Kb, Vb, Sc, AO, T, /*causal=*/1);
        resadd(stream, Y, T);

        // cross-attention over encoder output
        k_ln<<<BL, 256, 0, stream>>>(Y, a1, b1l, Hb, DMODEL);
        mha(stream, Hb, ENC, dec_cq + i * DD, dec_ck + i * DD, dec_cv + i * DD,
            dec_co + i * DD, Qb, Kb, Vb, Sc, AO, T, /*causal=*/0);
        resadd(stream, Y, T);

        // FFN
        k_ln<<<BL, 256, 0, stream>>>(Y, a2, b2l, Hb, DMODEL);
        gemm(stream, Hb, dec_w1 + i * DF, dec_b1 + (size_t)i * DFFN, FF,
             BL, DFFN, DMODEL, DMODEL, DFFN, DFFN,
             1, 1, 0, 0, 0, 0, 0, 0, 1.f, /*relu=*/1, 0);
        gemm(stream, FF, dec_w2 + i * DF, dec_b2 + (size_t)i * DMODEL, T,
             BL, DMODEL, DFFN, DFFN, DMODEL, DMODEL,
             1, 1, 0, 0, 0, 0, 0, 0, 1.f, 0, 0);
        resadd(stream, Y, T);
    }

    // ===================== final projection + log-softmax =====================
    k_ln<<<BL, 256, 0, stream>>>(Y, dec_fa, dec_fb, Hb, DMODEL);
    gemm(stream, Hb, proj_w, proj_b, (float*)d_out,
         BL, VTGT, DMODEL, DMODEL, VTGT, VTGT,
         1, 1, 0, 0, 0, 0, 0, 0, 1.f, 0, 0);
    k_logsoftmax<<<BL, 256, 0, stream>>>((float*)d_out, VTGT);
}
